// MultiHeadAttentionClassical_65481071399994
// MI455X (gfx1250) — matrix operand order repaired and checked by reading, not compiled, not run on hardware
//
#include <hip/hip_runtime.h>
#include <hip/hip_bf16.h>
#include <stdint.h>

typedef __attribute__((ext_vector_type(16))) _Float16 v16h;
typedef __attribute__((ext_vector_type(8)))  _Float16 v8h;
typedef __attribute__((ext_vector_type(4)))  _Float16 v4h;
typedef __attribute__((ext_vector_type(8)))  float    v8f;
typedef __attribute__((ext_vector_type(4)))  int      v4i;

#define BATCH 2
#define SEQ   2048
#define DIM   1024
#define HEADS 16
#define HDIM  64
#define MTOT  (BATCH * SEQ)   // 4096

__device__ __forceinline__ v8f wmma16x16x32(v16h a, v16h b, v8f c) {
    return __builtin_amdgcn_wmma_f32_16x16x32_f16(
        /*neg_a=*/false, a, /*neg_b=*/false, b,
        /*c_mod=*/(short)0, c, /*reuse_a=*/false, /*reuse_b=*/false);
}

// ---------------------------------------------------------------------------
// gfx1250 async global->LDS staging (ASYNCcnt path), sync fallback.
// ---------------------------------------------------------------------------
#if defined(__has_builtin)
#if __has_builtin(__builtin_amdgcn_global_load_async_to_lds_b128)
#define USE_ASYNC_LDS 1
#endif
#endif

typedef __attribute__((address_space(1))) v4i* g_v4i_ptr;
typedef __attribute__((address_space(3))) v4i* l_v4i_ptr;

__device__ __forceinline__ void stage_b128(const _Float16* g, _Float16* l) {
#ifdef USE_ASYNC_LDS
    __builtin_amdgcn_global_load_async_to_lds_b128(
        (g_v4i_ptr)(uintptr_t)g,
        (l_v4i_ptr)(uint32_t)(uintptr_t)l,
        0, 0);
#else
    *(v8h*)l = *(const v8h*)g;
#endif
}

__device__ __forceinline__ void wait_stage() {
#ifdef USE_ASYNC_LDS
#if __has_builtin(__builtin_amdgcn_s_wait_asynccnt)
    __builtin_amdgcn_s_wait_asynccnt(0);
#else
    asm volatile("s_wait_asynccnt 0x0" ::: "memory");
#endif
#endif
}

// ---------------------------------------------------------------------------
// fp32 -> f16 conversion, float4 vectorized
// ---------------------------------------------------------------------------
__global__ __launch_bounds__(256)
void cvt_f32_f16(const float4* __restrict__ src, v4h* __restrict__ dst, int n4) {
    for (int i = blockIdx.x * blockDim.x + threadIdx.x; i < n4;
         i += gridDim.x * blockDim.x) {
        float4 f = src[i];
        v4h h = { (_Float16)f.x, (_Float16)f.y, (_Float16)f.z, (_Float16)f.w };
        dst[i] = h;
    }
}

// ---------------------------------------------------------------------------
// GEMM: C[M,N] = A[M,K] * Bw[N,K]^T + bias[N]
// Block tile 256x128, 8 waves (4x2), wave tile 64x64 (4x4 WMMA C-tiles).
// Double-buffered LDS: tile i+1 staged async while tile i is computed.
// ---------------------------------------------------------------------------
#define GBM 256
#define GBN 128
#define GBK 64

template <typename OutT>
__global__ __launch_bounds__(256)
void gemm_bias(const _Float16* __restrict__ A, const _Float16* __restrict__ Bw,
               const float* __restrict__ bias, OutT* __restrict__ C,
               int M, int N, int K) {
    __shared__ alignas(16) _Float16 a_s[2][GBM][GBK];   // 64 KB
    __shared__ alignas(16) _Float16 b_s[2][GBN][GBK];   // 32 KB

    const int tid   = threadIdx.x;
    const int lane  = tid & 31;
    const int wave  = tid >> 5;          // 0..7
    const int wm    = (wave & 3) * 64;   // 4 waves along M
    const int wn    = (wave >> 2) * 64;  // 2 waves along N
    const int bm    = blockIdx.x * GBM;
    const int bn    = blockIdx.y * GBN;
    const int lmod  = lane & 15;
    const int lhalf = lane >> 4;

    v8f acc[4][4] = {};

    const _Float16* ga = A  + (size_t)(bm + tid) * K;          // row per thread
    const _Float16* gb = Bw + (size_t)(bn + (tid >> 1)) * K + (tid & 1) * 32;

    auto stage_tile = [&](int k0, int buf) {
#pragma unroll
        for (int j = 0; j < 8; ++j)
            stage_b128(ga + k0 + j * 8, &a_s[buf][tid][j * 8]);
#pragma unroll
        for (int j = 0; j < 4; ++j)
            stage_b128(gb + k0 + j * 8, &b_s[buf][tid >> 1][(tid & 1) * 32 + j * 8]);
        if (k0 + GBK < K) {   // L2 prefetch one more tile ahead
            __builtin_prefetch(ga + k0 + GBK, 0, 3);
            __builtin_prefetch(gb + k0 + GBK, 0, 3);
        }
    };

    const int nk = K / GBK;
    stage_tile(0, 0);
    wait_stage();
    __syncthreads();

    for (int ik = 0; ik < nk; ++ik) {
        const int cur = ik & 1;
        if (ik + 1 < nk)
            stage_tile((ik + 1) * GBK, cur ^ 1);   // overlap with compute below

#pragma unroll
        for (int kk = 0; kk < GBK; kk += 32) {
            v16h af[4], bf[4];
#pragma unroll
            for (int i = 0; i < 4; ++i) {
                // A-layout: lane m holds M=m, K = lhalf*8+[0..7] and 16+lhalf*8+[0..7]
                int row = wm + i * 16 + lmod;
                int kb  = kk + lhalf * 8;
                ((v8h*)&af[i])[0] = *(const v8h*)&a_s[cur][row][kb];
                ((v8h*)&af[i])[1] = *(const v8h*)&a_s[cur][row][kb + 16];
            }
#pragma unroll
            for (int i = 0; i < 4; ++i) {
                // B-layout: lane n holds N=n, K = lhalf*8+[0..7] and 16+lhalf*8+[0..7]
                int col = wn + i * 16 + lmod;
                int kb  = kk + lhalf * 8;
                ((v8h*)&bf[i])[0] = *(const v8h*)&b_s[cur][col][kb];
                ((v8h*)&bf[i])[1] = *(const v8h*)&b_s[cur][col][kb + 16];
            }
#pragma unroll
            for (int mi = 0; mi < 4; ++mi)
#pragma unroll
                for (int ni = 0; ni < 4; ++ni)
                    acc[mi][ni] = wmma16x16x32(af[mi], bf[ni], acc[mi][ni]);
        }
        wait_stage();      // next tile's async transfers done (hidden by WMMAs)
        __syncthreads();
    }

    // epilogue: lane holds fixed column, rows r + 8*lhalf
#pragma unroll
    for (int mi = 0; mi < 4; ++mi)
#pragma unroll
        for (int ni = 0; ni < 4; ++ni) {
            int col  = bn + wn + ni * 16 + lmod;
            float bv = bias[col];
#pragma unroll
            for (int r = 0; r < 8; ++r) {
                int row = bm + wm + mi * 16 + r + lhalf * 8;
                C[(size_t)row * N + col] = (OutT)(acc[mi][ni][r] + bv);
            }
        }
}

// ---------------------------------------------------------------------------
// Flash attention: grid (S/128, H, B); 8 waves, each wave owns 16 q-rows.
// 64-key tiles, double-buffered K/V staging overlapped with WMMA + softmax.
// ---------------------------------------------------------------------------
#define KT 64

__global__ __launch_bounds__(256)
void flash_attn(const _Float16* __restrict__ Q, const _Float16* __restrict__ Km,
                const _Float16* __restrict__ Vm, const int* __restrict__ mask,
                _Float16* __restrict__ O) {
    __shared__ alignas(16) _Float16 k_s[2][KT][HDIM];    // 16 KB, row-major keys
    __shared__ alignas(16) _Float16 vt_s[2][HDIM][KT];   // 16 KB, V transposed
    __shared__ alignas(16) _Float16 p_s[8][16][KT];      // 16 KB, per-wave P buffer

    const int tid   = threadIdx.x;
    const int lane  = tid & 31;
    const int wave  = tid >> 5;
    const int lmod  = lane & 15;
    const int lhalf = lane >> 4;
    const int b     = blockIdx.z;
    const int h     = blockIdx.y;
    const int q0    = blockIdx.x * 128 + wave * 16;
    const size_t base = (size_t)b * SEQ * DIM + (size_t)h * HDIM; // row stride DIM

    // Q fragments: 16x64 = two 16x32 A-tiles (chained over head dim)
    v16h qf[2];
    {
        const _Float16* qp = Q + base + (size_t)(q0 + lmod) * DIM;
#pragma unroll
        for (int c = 0; c < 2; ++c) {
            int kb = c * 32 + lhalf * 8;
            ((v8h*)&qf[c])[0] = *(const v8h*)(qp + kb);
            ((v8h*)&qf[c])[1] = *(const v8h*)(qp + kb + 16);
        }
    }

    v8f o[4] = {};                 // 16 x 64 output accumulator (4 d-tiles)
    float mrow[8], lrow[8];
#pragma unroll
    for (int r = 0; r < 8; ++r) { mrow[r] = -1e30f; lrow[r] = 0.0f; }
    const float scale = 0.125f;    // 1/sqrt(64)

    // staging coords: 64 keys x 64 d; 4 threads per key, 16 halfs each
    const int skey = tid >> 2;
    const int sd0  = (tid & 3) * 16;

    auto stage_kv = [&](int kt, int buf) {
        const _Float16* kp = Km + base + (size_t)(kt + skey) * DIM + sd0;
        stage_b128(kp,     &k_s[buf][skey][sd0]);       // async K
        stage_b128(kp + 8, &k_s[buf][skey][sd0 + 8]);
        const _Float16* vp = Vm + base + (size_t)(kt + skey) * DIM + sd0;
        v8h vv0 = *(const v8h*)vp;                      // V: load + transpose scatter
        v8h vv1 = *(const v8h*)(vp + 8);
#pragma unroll
        for (int j = 0; j < 8; ++j) {
            vt_s[buf][sd0 + j][skey]     = vv0[j];
            vt_s[buf][sd0 + 8 + j][skey] = vv1[j];
        }
        if (kt + KT < SEQ) {
            __builtin_prefetch(kp + (size_t)KT * DIM, 0, 3);
            __builtin_prefetch(vp + (size_t)KT * DIM, 0, 3);
        }
    };

    const int niter = SEQ / KT;
    stage_kv(0, 0);
    wait_stage();
    __syncthreads();

    for (int it = 0; it < niter; ++it) {
        const int kt  = it * KT;
        const int cur = it & 1;
        if (it + 1 < niter)
            stage_kv(kt + KT, cur ^ 1);    // overlap next tile with compute

        // scores S = Q * K^T : 4 key-column tiles x 2-step K-chain
        v8f sc[4] = {};
#pragma unroll
        for (int nt = 0; nt < 4; ++nt) {
#pragma unroll
            for (int c = 0; c < 2; ++c) {
                v16h bf;
                int key = nt * 16 + lmod;
                int kb  = c * 32 + lhalf * 8;
                ((v8h*)&bf)[0] = *(const v8h*)&k_s[cur][key][kb];
                ((v8h*)&bf)[1] = *(const v8h*)&k_s[cur][key][kb + 16];
                sc[nt] = wmma16x16x32(qf[c], bf, sc[nt]);
            }
        }

        // scale + mask (lane's C-tile column is a fixed key)
        float madd[4];
#pragma unroll
        for (int nt = 0; nt < 4; ++nt)
            madd[nt] = mask[b * SEQ + kt + nt * 16 + lmod] ? 0.0f : -1e9f;
#pragma unroll
        for (int nt = 0; nt < 4; ++nt)
#pragma unroll
            for (int r = 0; r < 8; ++r)
                sc[nt][r] = sc[nt][r] * scale + madd[nt];

        // online softmax: row max / rescale / exp / row sum
#pragma unroll
        for (int r = 0; r < 8; ++r) {
            float v = fmaxf(fmaxf(sc[0][r], sc[1][r]), fmaxf(sc[2][r], sc[3][r]));
#pragma unroll
            for (int off = 8; off >= 1; off >>= 1)
                v = fmaxf(v, __shfl_xor(v, off, 32));
            float mnew  = fmaxf(mrow[r], v);
            float alpha = __expf(mrow[r] - mnew);
            mrow[r] = mnew;
            lrow[r] *= alpha;
#pragma unroll
            for (int t = 0; t < 4; ++t) o[t][r] *= alpha;
            float s = 0.0f;
#pragma unroll
            for (int nt = 0; nt < 4; ++nt) {
                float p = __expf(sc[nt][r] - mnew);
                sc[nt][r] = p;
                s += p;
            }
#pragma unroll
            for (int off = 8; off >= 1; off >>= 1)
                s += __shfl_xor(s, off, 32);
            lrow[r] += s;
        }

        // transpose P (C-layout fp32) -> row-major f16 via wave-private LDS
#pragma unroll
        for (int nt = 0; nt < 4; ++nt)
#pragma unroll
            for (int r = 0; r < 8; ++r)
                p_s[wave][r + lhalf * 8][nt * 16 + lmod] = (_Float16)sc[nt][r];
        // same-wave LDS ops are in-order; compiler inserts s_wait_dscnt

        // O += P * V : P as A-fragments (64-key chain), V^T rows as B-fragments
        v16h pf[2];
#pragma unroll
        for (int c = 0; c < 2; ++c) {
            int kb = c * 32 + lhalf * 8;
            ((v8h*)&pf[c])[0] = *(const v8h*)&p_s[wave][lmod][kb];
            ((v8h*)&pf[c])[1] = *(const v8h*)&p_s[wave][lmod][kb + 16];
        }
#pragma unroll
        for (int nt = 0; nt < 4; ++nt) {
#pragma unroll
            for (int c = 0; c < 2; ++c) {
                v16h bf;
                int d  = nt * 16 + lmod;
                int kb = c * 32 + lhalf * 8;
                ((v8h*)&bf)[0] = *(const v8h*)&vt_s[cur][d][kb];
                ((v8h*)&bf)[1] = *(const v8h*)&vt_s[cur][d][kb + 16];
                o[nt] = wmma16x16x32(pf[c], bf, o[nt]);
            }
        }
        wait_stage();      // next K tile's async transfers done (hidden by compute)
        __syncthreads();
    }

    // normalize and store (f16, same [B,S,H*Hd] layout)
#pragma unroll
    for (int nt = 0; nt < 4; ++nt) {
        int d = nt * 16 + lmod;
#pragma unroll
        for (int r = 0; r < 8; ++r) {
            int row = q0 + r + lhalf * 8;
            O[base + (size_t)row * DIM + d] = (_Float16)(o[nt][r] / lrow[r]);
        }
    }
}

// ---------------------------------------------------------------------------
// Launch
// ---------------------------------------------------------------------------
extern "C" void kernel_launch(void* const* d_in, const int* in_sizes, int n_in,
                              void* d_out, int out_size, void* d_ws, size_t ws_size,
                              hipStream_t stream) {
    const float* x    = (const float*)d_in[0];
    const int*   mask = (const int*)d_in[1];
    const float* Wq   = (const float*)d_in[2];
    const float* bq   = (const float*)d_in[3];
    const float* Wk   = (const float*)d_in[4];
    const float* bk   = (const float*)d_in[5];
    const float* Wv   = (const float*)d_in[6];
    const float* bv   = (const float*)d_in[7];
    const float* Wo   = (const float*)d_in[8];
    const float* bo   = (const float*)d_in[9];

    const size_t nX = (size_t)MTOT * DIM;   // 4M elements
    const size_t nW = (size_t)DIM * DIM;    // 1M elements

    _Float16* xh  = (_Float16*)d_ws;
    _Float16* wqh = xh  + nX;
    _Float16* wkh = wqh + nW;
    _Float16* wvh = wkh + nW;
    _Float16* woh = wvh + nW;
    _Float16* qh  = woh + nW;
    _Float16* kh  = qh  + nX;
    _Float16* vh  = kh  + nX;
    _Float16* ah  = vh  + nX;

    // fp32 -> f16 conversions (float4 vectorized)
    cvt_f32_f16<<<1024, 256, 0, stream>>>((const float4*)x,  (v4h*)xh,  (int)(nX / 4));
    cvt_f32_f16<<<512,  256, 0, stream>>>((const float4*)Wq, (v4h*)wqh, (int)(nW / 4));
    cvt_f32_f16<<<512,  256, 0, stream>>>((const float4*)Wk, (v4h*)wkh, (int)(nW / 4));
    cvt_f32_f16<<<512,  256, 0, stream>>>((const float4*)Wv, (v4h*)wvh, (int)(nW / 4));
    cvt_f32_f16<<<512,  256, 0, stream>>>((const float4*)Wo, (v4h*)woh, (int)(nW / 4));

    // Q/K/V projections: [4096,1024] = x[4096,1024] * W[1024,1024]^T + b
    dim3 ggrid(MTOT / GBM, DIM / GBN);   // 16 x 8
    gemm_bias<_Float16><<<ggrid, 256, 0, stream>>>(xh, wqh, bq, qh, MTOT, DIM, DIM);
    gemm_bias<_Float16><<<ggrid, 256, 0, stream>>>(xh, wkh, bk, kh, MTOT, DIM, DIM);
    gemm_bias<_Float16><<<ggrid, 256, 0, stream>>>(xh, wvh, bv, vh, MTOT, DIM, DIM);

    // flash attention
    dim3 fgrid(SEQ / 128, HEADS, BATCH);
    flash_attn<<<fgrid, 256, 0, stream>>>(qh, kh, vh, mask, ah);

    // output projection (fp32 out)
    gemm_bias<float><<<ggrid, 256, 0, stream>>>(ah, woh, bo, (float*)d_out, MTOT, DIM, DIM);
}
